// PointNetFeaturePropagation_1726576857085
// MI455X (gfx1250) — matrix operand order repaired and checked by reading, compiled, not run on hardware
//
#include <hip/hip_runtime.h>
#include <hip/hip_bf16.h>

// ---------------------------------------------------------------------------
// PointNet++ Feature Propagation for MI455X (gfx1250, wave32, WMMA)
//   B=8, N=8192, S=2048, D=256, C=512.
//   GEMMs (2 x 65536x512x512 = 68.7 GFLOP) run on v_wmma_f32_16x16x32_f16
//   with a double-buffered k-loop; wave tile 32x64 (8 accumulators).
// ---------------------------------------------------------------------------

typedef _Float16 v16h __attribute__((ext_vector_type(16)));
typedef _Float16 v8h  __attribute__((ext_vector_type(8)));
typedef float    v8f  __attribute__((ext_vector_type(8)));

#define B_   8
#define N_   8192
#define S_   2048
#define D_   256
#define C_   512
#define M_   (B_ * N_)          // 65536 rows
#define NCHUNK_ 64              // stat reduction chunks (fixed order -> deterministic)

// ---------------------------------------------------------------------------
// 1. 3-NN search. One block = 256 points of one batch; xyz2 + |q|^2 in LDS.
// ---------------------------------------------------------------------------
__global__ __launch_bounds__(256) void knn3_kernel(
    const float* __restrict__ xyz1, const float* __restrict__ xyz2,
    int* __restrict__ idx, float* __restrict__ wgt) {
  __shared__ float sx[S_], sy[S_], sz[S_], sq[S_];
  const int b = blockIdx.y;
  const int t = threadIdx.x;
  const float* X2 = xyz2 + (size_t)b * 3 * S_;
  for (int i = t; i < S_; i += 256) {
    float xx = X2[i], yy = X2[S_ + i], zz = X2[2 * S_ + i];
    sx[i] = xx; sy[i] = yy; sz[i] = zz;
    sq[i] = xx * xx + yy * yy + zz * zz;
  }
  __syncthreads();

  const int n = blockIdx.x * 256 + t;
  const float* X1 = xyz1 + (size_t)b * 3 * N_;
  const float px = X1[n], py = X1[N_ + n], pz = X1[2 * N_ + n];
  const float pn = px * px + py * py + pz * pz;

  float d0 = 1e30f, d1 = 1e30f, d2 = 1e30f;
  int   i0 = 0, i1 = 0, i2 = 0;
  for (int s = 0; s < S_; ++s) {
    float dot = px * sx[s] + py * sy[s] + pz * sz[s];
    float d = pn + sq[s] - 2.0f * dot;   // same formula as reference
    if (d < d2) {
      if (d < d1) {
        if (d < d0) { d2 = d1; i2 = i1; d1 = d0; i1 = i0; d0 = d; i0 = s; }
        else        { d2 = d1; i2 = i1; d1 = d;  i1 = s; }
      } else        { d2 = d;  i2 = s; }
    }
  }
  float r0 = 1.0f / (d0 + 1e-8f);
  float r1 = 1.0f / (d1 + 1e-8f);
  float r2 = 1.0f / (d2 + 1e-8f);
  float inv = 1.0f / (r0 + r1 + r2);
  size_t m = (size_t)b * N_ + n;
  idx[m * 3 + 0] = i0; idx[m * 3 + 1] = i1; idx[m * 3 + 2] = i2;
  wgt[m * 3 + 0] = r0 * inv; wgt[m * 3 + 1] = r1 * inv; wgt[m * 3 + 2] = r2 * inv;
}

// ---------------------------------------------------------------------------
// 2. Interpolate + concat -> f16 GEMM input, row-major [M, 512].
//    One block per point; thread t = channel. Per-batch points1/points2
//    (8MB + 2MB) are L2-resident (192MB L2) so strided/gather reads are cheap.
// ---------------------------------------------------------------------------
__global__ __launch_bounds__(256) void interp_concat_kernel(
    const float* __restrict__ points1, const float* __restrict__ points2,
    const int* __restrict__ idx, const float* __restrict__ wgt,
    _Float16* __restrict__ A) {
  const size_t m = blockIdx.x;                // 0..65535
  const int b = (int)(m >> 13);               // m / 8192
  const int n = (int)(m & (N_ - 1));
  const int t = threadIdx.x;                  // channel 0..255
  const int j0 = idx[m * 3 + 0], j1 = idx[m * 3 + 1], j2 = idx[m * 3 + 2];
  const float w0 = wgt[m * 3 + 0], w1 = wgt[m * 3 + 1], w2 = wgt[m * 3 + 2];
  const float* P1 = points1 + ((size_t)b * D_ + t) * N_;
  const float* P2 = points2 + ((size_t)b * D_ + t) * S_;
  float v1 = P1[n];
  float vi = w0 * P2[j0] + w1 * P2[j1] + w2 * P2[j2];
  _Float16* row = A + m * C_;
  row[t]      = (_Float16)v1;   // cols   0..255 : skip features
  row[D_ + t] = (_Float16)vi;   // cols 256..511 : interpolated
}

// ---------------------------------------------------------------------------
// 3. fp32 -> f16 weight conversion.
// ---------------------------------------------------------------------------
__global__ void w_to_f16_kernel(const float* __restrict__ in,
                                _Float16* __restrict__ out, int n) {
  int i = blockIdx.x * blockDim.x + threadIdx.x;
  if (i < n) out[i] = (_Float16)in[i];
}

// ---------------------------------------------------------------------------
// 4. WMMA GEMM: C[M,512] = A[M,512] @ W[512,512]^T + bias.
//    A row-major f16, W row-major [O,K] f16 -> per-lane fragments are
//    contiguous loads, k = 16*half + s in both A and B (no LDS needed).
//    Wave tile 32x64 (8 accumulators), double-buffered k-loop so loads for
//    chunk k+32 are in flight while the 8 WMMAs of chunk k execute.
//    Block = 8 waves stacked in M -> 256x64 tile.
// ---------------------------------------------------------------------------
__device__ __forceinline__ v16h load_a_frag(const _Float16* __restrict__ Arow,
                                            int ka) {
  v8h lo = *(const v8h*)(Arow + ka);
  v8h hi = *(const v8h*)(Arow + ka + 8);
  v16h a;
#pragma unroll
  for (int i = 0; i < 8; ++i) { a[i] = lo[i]; a[8 + i] = hi[i]; }
  return a;
}

__global__ __launch_bounds__(256) void gemm_f16_kernel(
    const _Float16* __restrict__ A, const _Float16* __restrict__ W,
    const float* __restrict__ bias, float* __restrict__ C) {
  const int K = C_;
  const int lane = threadIdx.x & 31;
  const int wave = threadIdx.x >> 5;
  const int m0 = blockIdx.x * 256 + wave * 32;   // 32 rows per wave
  const int n0 = blockIdx.y * 64;                // 64 cols per wave
  const int half = lane >> 4;                    // 0 | 1
  const int l16  = lane & 15;

  // A fragment rows (two 16-row blocks), per-lane base pointers.
  const _Float16* Ar0 = A + (size_t)(m0 + l16) * K + half * 16;
  const _Float16* Ar1 = A + (size_t)(m0 + 16 + l16) * K + half * 16;
  // B fragment rows of W (col n = weight row), 16 contiguous k at kc+16*half.
  const _Float16* Wr0 = W + (size_t)(n0 +  0 + l16) * K + half * 16;
  const _Float16* Wr1 = W + (size_t)(n0 + 16 + l16) * K + half * 16;
  const _Float16* Wr2 = W + (size_t)(n0 + 32 + l16) * K + half * 16;
  const _Float16* Wr3 = W + (size_t)(n0 + 48 + l16) * K + half * 16;

  v8f acc[2][4];
#pragma unroll
  for (int r = 0; r < 2; ++r)
#pragma unroll
    for (int j = 0; j < 4; ++j) acc[r][j] = v8f{};

  // Double-buffered fragments (fully unrolled loop -> pure SSA registers).
  v16h a0[2], a1[2], b0[2], b1[2], b2[2], b3[2];

  // Prologue: chunk 0 into slot 0.
  a0[0] = load_a_frag(Ar0, 0);
  a1[0] = load_a_frag(Ar1, 0);
  b0[0] = *(const v16h*)(Wr0);
  b1[0] = *(const v16h*)(Wr1);
  b2[0] = *(const v16h*)(Wr2);
  b3[0] = *(const v16h*)(Wr3);

  const int NITER = K / 32;   // 16
#pragma unroll
  for (int it = 0; it < NITER; ++it) {
    const int cur = it & 1;
    const int nxt = cur ^ 1;
    if (it + 1 < NITER) {
      const int kc = (it + 1) * 32;
      a0[nxt] = load_a_frag(Ar0, kc);
      a1[nxt] = load_a_frag(Ar1, kc);
      b0[nxt] = *(const v16h*)(Wr0 + kc);
      b1[nxt] = *(const v16h*)(Wr1 + kc);
      b2[nxt] = *(const v16h*)(Wr2 + kc);
      b3[nxt] = *(const v16h*)(Wr3 + kc);
    }
    acc[0][0] = __builtin_amdgcn_wmma_f32_16x16x32_f16(false, a0[cur], false, b0[cur],
                                                       (short)0, acc[0][0], false, false);
    acc[0][1] = __builtin_amdgcn_wmma_f32_16x16x32_f16(false, a0[cur], false, b1[cur],
                                                       (short)0, acc[0][1], false, false);
    acc[0][2] = __builtin_amdgcn_wmma_f32_16x16x32_f16(false, a0[cur], false, b2[cur],
                                                       (short)0, acc[0][2], false, false);
    acc[0][3] = __builtin_amdgcn_wmma_f32_16x16x32_f16(false, a0[cur], false, b3[cur],
                                                       (short)0, acc[0][3], false, false);
    acc[1][0] = __builtin_amdgcn_wmma_f32_16x16x32_f16(false, a1[cur], false, b0[cur],
                                                       (short)0, acc[1][0], false, false);
    acc[1][1] = __builtin_amdgcn_wmma_f32_16x16x32_f16(false, a1[cur], false, b1[cur],
                                                       (short)0, acc[1][1], false, false);
    acc[1][2] = __builtin_amdgcn_wmma_f32_16x16x32_f16(false, a1[cur], false, b2[cur],
                                                       (short)0, acc[1][2], false, false);
    acc[1][3] = __builtin_amdgcn_wmma_f32_16x16x32_f16(false, a1[cur], false, b3[cur],
                                                       (short)0, acc[1][3], false, false);
  }

  // C/D layout: VGPR v, lanes 0-15 -> row base+v, lanes 16-31 -> row base+8+v;
  // col = l16. Fuse bias add into the store epilogue.
#pragma unroll
  for (int r = 0; r < 2; ++r) {
    const int mbase = m0 + r * 16 + half * 8;
#pragma unroll
    for (int j = 0; j < 4; ++j) {
      const int n = n0 + j * 16 + l16;
      const float bv = bias[n];
#pragma unroll
      for (int v = 0; v < 8; ++v) {
        C[(size_t)(mbase + v) * C_ + n] = acc[r][j][v] + bv;
      }
    }
  }
}

// ---------------------------------------------------------------------------
// 5. Deterministic BN batch stats (sum / sumsq), two fixed-order stages.
// ---------------------------------------------------------------------------
__global__ __launch_bounds__(512) void col_partial_kernel(
    const float* __restrict__ H, float* __restrict__ partial) {
  const int c = threadIdx.x;            // 0..511 (one column per thread)
  const int chunk = blockIdx.x;         // 0..63
  const int rows = M_ / NCHUNK_;        // 1024
  const float* p = H + ((size_t)chunk * rows) * C_ + c;
  float s = 0.0f, s2 = 0.0f;
  for (int r = 0; r < rows; ++r) {
    float v = p[(size_t)r * C_];
    s += v; s2 += v * v;
  }
  partial[chunk * C_ + c] = s;
  partial[NCHUNK_ * C_ + chunk * C_ + c] = s2;
}

__global__ __launch_bounds__(512) void col_finalize_kernel(
    const float* __restrict__ partial, const float* __restrict__ g,
    const float* __restrict__ be, float* __restrict__ scale,
    float* __restrict__ shift) {
  const int c = threadIdx.x;
  float s = 0.0f, s2 = 0.0f;
  for (int ch = 0; ch < NCHUNK_; ++ch) {          // fixed order -> deterministic
    s  += partial[ch * C_ + c];
    s2 += partial[NCHUNK_ * C_ + ch * C_ + c];
  }
  const float mean = s * (1.0f / (float)M_);
  const float var  = s2 * (1.0f / (float)M_) - mean * mean;  // biased, as jnp.var
  const float sc = g[c] * rsqrtf(var + 1e-5f);
  scale[c] = sc;
  shift[c] = be[c] - mean * sc;
}

// ---------------------------------------------------------------------------
// 6. BN + ReLU -> f16 (input to second GEMM).
// ---------------------------------------------------------------------------
__global__ __launch_bounds__(256) void bn_relu_to_f16_kernel(
    const float* __restrict__ H, const float* __restrict__ scale,
    const float* __restrict__ shift, _Float16* __restrict__ O) {
  const size_t base = ((size_t)blockIdx.x * 256 + threadIdx.x) * 4;
#pragma unroll
  for (int k = 0; k < 4; ++k) {
    size_t i = base + k;
    int c = (int)(i & (C_ - 1));
    float v = H[i] * scale[c] + shift[c];
    O[i] = (_Float16)fmaxf(v, 0.0f);
  }
}

// ---------------------------------------------------------------------------
// 7. BN + ReLU + transpose [B*N,512] -> [B,512,N] (32x32 LDS tile).
// ---------------------------------------------------------------------------
__global__ __launch_bounds__(256) void bn_relu_transpose_kernel(
    const float* __restrict__ H, const float* __restrict__ scale,
    const float* __restrict__ shift, float* __restrict__ out) {
  __shared__ float tile[32][33];
  const int b  = blockIdx.z;
  const int c0 = blockIdx.y * 32;
  const int n0 = blockIdx.x * 32;
  const int tx = threadIdx.x & 31;
  const int ty = threadIdx.x >> 5;
  for (int r = ty; r < 32; r += 8) {
    const int n = n0 + r, c = c0 + tx;
    float v = H[((size_t)b * N_ + n) * C_ + c] * scale[c] + shift[c];
    tile[r][tx] = fmaxf(v, 0.0f);
  }
  __syncthreads();
  for (int r = ty; r < 32; r += 8) {
    const int c = c0 + r, n = n0 + tx;
    out[((size_t)b * C_ + c) * N_ + n] = tile[tx][r];
  }
}

// ---------------------------------------------------------------------------
// Launch
// ---------------------------------------------------------------------------
extern "C" void kernel_launch(void* const* d_in, const int* in_sizes, int n_in,
                              void* d_out, int out_size, void* d_ws, size_t ws_size,
                              hipStream_t stream) {
  const float* xyz1 = (const float*)d_in[0];
  const float* xyz2 = (const float*)d_in[1];
  const float* pts1 = (const float*)d_in[2];
  const float* pts2 = (const float*)d_in[3];
  const float* W1   = (const float*)d_in[4];
  const float* b1   = (const float*)d_in[5];
  const float* g1   = (const float*)d_in[6];
  const float* be1  = (const float*)d_in[7];
  const float* W2   = (const float*)d_in[8];
  const float* b2   = (const float*)d_in[9];
  const float* g2   = (const float*)d_in[10];
  const float* be2  = (const float*)d_in[11];
  float* out = (float*)d_out;

  // Workspace layout (bytes)
  char* ws = (char*)d_ws;
  _Float16* A16   = (_Float16*)(ws);                                   //  64 MB
  float*    H32   = (float*)(ws + (size_t)M_ * C_ * 2);                // 128 MB
  char* p = ws + (size_t)M_ * C_ * 2 + (size_t)M_ * C_ * 4;
  _Float16* W1h   = (_Float16*)p;            p += (size_t)C_ * C_ * 2;
  _Float16* W2h   = (_Float16*)p;            p += (size_t)C_ * C_ * 2;
  float*    part  = (float*)p;               p += (size_t)2 * NCHUNK_ * C_ * 4;
  float*    scal  = (float*)p;               p += (size_t)C_ * 4;
  float*    shft  = (float*)p;               p += (size_t)C_ * 4;
  int*      knn_i = (int*)p;                 p += (size_t)M_ * 3 * 4;
  float*    knn_w = (float*)p;               p += (size_t)M_ * 3 * 4;

  // 1. 3-NN
  knn3_kernel<<<dim3(N_ / 256, B_), 256, 0, stream>>>(xyz1, xyz2, knn_i, knn_w);

  // 2. interpolate + concat -> f16 GEMM input
  interp_concat_kernel<<<dim3(M_), 256, 0, stream>>>(pts1, pts2, knn_i, knn_w, A16);

  // 3. weights to f16
  w_to_f16_kernel<<<dim3((C_ * C_) / 256), 256, 0, stream>>>(W1, W1h, C_ * C_);
  w_to_f16_kernel<<<dim3((C_ * C_) / 256), 256, 0, stream>>>(W2, W2h, C_ * C_);

  const dim3 gemm_grid(M_ / 256, C_ / 64);

  // 4. layer 1: GEMM + bias -> stats -> BN+ReLU -> f16
  gemm_f16_kernel<<<gemm_grid, 256, 0, stream>>>(A16, W1h, b1, H32);
  col_partial_kernel<<<dim3(NCHUNK_), 512, 0, stream>>>(H32, part);
  col_finalize_kernel<<<dim3(1), 512, 0, stream>>>(part, g1, be1, scal, shft);
  bn_relu_to_f16_kernel<<<dim3((size_t)M_ * C_ / (256 * 4)), 256, 0, stream>>>(
      H32, scal, shft, A16);

  // 5. layer 2: GEMM + bias -> stats -> BN+ReLU + transpose to [B,512,N]
  gemm_f16_kernel<<<gemm_grid, 256, 0, stream>>>(A16, W2h, b2, H32);
  col_partial_kernel<<<dim3(NCHUNK_), 512, 0, stream>>>(H32, part);
  col_finalize_kernel<<<dim3(1), 512, 0, stream>>>(part, g2, be2, scal, shft);
  bn_relu_transpose_kernel<<<dim3(N_ / 32, C_ / 32, B_), 256, 0, stream>>>(
      H32, scal, shft, out);

  (void)in_sizes; (void)n_in; (void)out_size; (void)ws_size;
}
